// StackedGates_18519898980948
// MI455X (gfx1250) — compile-verified, matrix operand order checked by reading, not run on hardware
//
#include <hip/hip_runtime.h>
#include <hip/hip_bf16.h>

typedef __attribute__((ext_vector_type(16))) _Float16 v16h;
typedef __attribute__((ext_vector_type(8)))  _Float16 v8h;
typedef __attribute__((ext_vector_type(8)))  float    v8f;
typedef __attribute__((ext_vector_type(4)))  unsigned int u32x4;
typedef __attribute__((ext_vector_type(4)))  int      i32x4;
typedef __attribute__((ext_vector_type(8)))  int      i32x8;

#define TSEQ 8192
#define NTOT 65536            // B*T = 8*8192

// Tensor Data Mover availability (toolchain-dependent arity; guard hard).
#if defined(__has_builtin)
#if __has_builtin(__builtin_amdgcn_tensor_load_to_lds) && \
    __has_builtin(__builtin_amdgcn_s_wait_tensorcnt)
#define USE_TDM 1
#endif
#endif
#ifndef USE_TDM
#define USE_TDM 0
#endif

// ---------------------------------------------------------------------------
// Weight prep: pack (O=128, I=128, K=2) f32 conv weight into f16 A = [W0 | W1]
// laid out [co][256] so the GEMM contracts over [x(t-d); x(t)].
// ---------------------------------------------------------------------------
__global__ __launch_bounds__(256) void prep_w2(const float* __restrict__ W,
                                               _Float16* __restrict__ A) {
    int id = blockIdx.x * 256 + threadIdx.x;       // 128*128 threads
    if (id >= 128 * 128) return;
    int co = id >> 7, ci = id & 127;
    A[co * 256 + ci]       = (_Float16)W[(co * 128 + ci) * 2 + 0];
    A[co * 256 + 128 + ci] = (_Float16)W[(co * 128 + ci) * 2 + 1];
}

__global__ __launch_bounds__(256) void prep_w1(const float* __restrict__ W,
                                               _Float16* __restrict__ A) {
    int id = blockIdx.x * 256 + threadIdx.x;       // 128*128 threads (K=1)
    if (id >= 128 * 128) return;
    A[id] = (_Float16)W[id];
}

// ---------------------------------------------------------------------------
// Front convs (tiny FLOPs). Activations stored TIME-MAJOR: act[n*C + ci].
// Pad-after semantics: y at t==0 is exactly 0.
// ---------------------------------------------------------------------------
__global__ __launch_bounds__(256) void front1_kernel(const float* __restrict__ x,
                                                     const float* __restrict__ w,
                                                     const float* __restrict__ b,
                                                     _Float16* __restrict__ c1) {
    int id = blockIdx.x * 256 + threadIdx.x;       // NTOT * 64
    if (id >= 64 * NTOT) return;
    int n = id >> 6, co = id & 63, t = n & (TSEQ - 1);
    float y = 0.f;
    if (t > 0) y = w[co * 2 + 0] * x[n - 1] + w[co * 2 + 1] * x[n] + b[co];
    c1[id] = (_Float16)y;                          // c1[n*64 + co]
}

__global__ __launch_bounds__(256) void front2_kernel(const float* __restrict__ w,
                                                     const float* __restrict__ b,
                                                     const _Float16* __restrict__ c1,
                                                     float* __restrict__ h,
                                                     _Float16* __restrict__ h16) {
    int id = blockIdx.x * 256 + threadIdx.x;       // NTOT * 128
    if (id >= 128 * NTOT) return;
    int n = id >> 7, co = id & 127, t = n & (TSEQ - 1);
    float acc = 0.f;
    if (t > 0) {
        const _Float16* c0 = c1 + (size_t)(n - 1) * 64;
        const _Float16* cc = c1 + (size_t)n * 64;
        for (int ci = 0; ci < 64; ++ci) {
            acc += w[(co * 64 + ci) * 2 + 0] * (float)c0[ci];
            acc += w[(co * 64 + ci) * 2 + 1] * (float)cc[ci];
        }
        acc += b[co];
    }
    h[id]   = acc;                                 // h[n*128 + co]  (f32)
    h16[id] = (_Float16)acc;
}

// ---------------------------------------------------------------------------
// Gate: z = tanh(f) * sigmoid(g)  (f16 in / f16 out, TRANS-heavy)
// ---------------------------------------------------------------------------
__global__ __launch_bounds__(256) void gate_kernel(const _Float16* __restrict__ f,
                                                   const _Float16* __restrict__ g,
                                                   _Float16* __restrict__ z) {
    int id = blockIdx.x * 256 + threadIdx.x;       // 128 * NTOT
    if (id >= 128 * NTOT) return;
    float fv = (float)f[id], gv = (float)g[id];
    float zv = tanhf(fv) * (1.f / (1.f + expf(-gv)));
    z[id] = (_Float16)zv;
}

// ---------------------------------------------------------------------------
// WMMA GEMM for one conv layer (activations time-major [n][128]).
//   KTOT=256, MODE=0 : Y16 = (f16)( [W0|W1] * [x(t-d);x(t)] + bias )   (chain)
//   KTOT=128, MODE=1 : hn = Wz + bias + h ; h=hn ; total+=hn ; h16=hn  (scale)
// WG = 256 threads = 8 wave32; WG tile 128(M) x 128(N); wave tile 16 x 128.
// A tile staged via Tensor Data Mover when available.
// ---------------------------------------------------------------------------
template <int KTOT, int MODE>
__global__ __launch_bounds__(256)
void gemm_conv_wmma(const _Float16* __restrict__ A,      // [128][KTOT] f16
                    const float*    __restrict__ bias,   // [128]
                    const _Float16* __restrict__ Xin,    // [n][128] f16
                    _Float16*       __restrict__ Yout16, // MODE 0: [n][128]
                    float*          __restrict__ Hres,   // MODE 1: [n][128]
                    float*          __restrict__ Tot,    // MODE 1: ref layout
                    _Float16*       __restrict__ H16out, // MODE 1: [n][128]
                    int d) {
    extern __shared__ _Float16 smem[];                   // dynamic LDS
    _Float16* Alds = smem;                               // 128*KTOT
    _Float16* Xlds = smem + 128 * KTOT;                  // 128*KTOT

    const int tid = threadIdx.x;
    const int n0  = blockIdx.x * 128;

    // ---- stage A (weights, L2-resident) into LDS via TDM ----
#if USE_TDM
    if (tid == 0) {
        const unsigned long long ga = (unsigned long long)(uintptr_t)A;
        const unsigned int       la = (unsigned int)(uintptr_t)Alds;
        u32x4 g0 = {0u, 0u, 0u, 0u};
        g0[0] = 1u;                                      // count=1, user mode
        g0[1] = la;                                      // lds_addr
        g0[2] = (unsigned int)ga;                        // global_addr[31:0]
        g0[3] = (unsigned int)((ga >> 32) & 0x01FFFFFFull) | (2u << 30); // +type=2
        i32x8 g1 = {0, 0, 0, 0, 0, 0, 0, 0};
        g1[0] = 1 << 16;                                 // data_size = 2 bytes
        g1[1] = (KTOT & 0xFFFF) << 16;                   // tensor_dim0 = KTOT
        g1[2] = 128 << 16;                               // tensor_dim1 = 128
        g1[3] = KTOT << 16;                              // tile_dim0   = KTOT
        g1[4] = 128;                                     // tile_dim1   = 128
        g1[5] = KTOT;                                    // tensor_dim0_stride
        i32x4 z4 = {0, 0, 0, 0};
#if __clang_major__ >= 23
        i32x8 z8 = {0, 0, 0, 0, 0, 0, 0, 0};
        __builtin_amdgcn_tensor_load_to_lds(g0, g1, z4, z4, z8, 0);
#else
        __builtin_amdgcn_tensor_load_to_lds(g0, g1, z4, z4, 0);
#endif
        __builtin_amdgcn_s_wait_tensorcnt(0);
    }
#else
    {
        const uint4* src = (const uint4*)A;
        uint4*       dst = (uint4*)Alds;
        const int total = (128 * KTOT) / 8;              // uint4 = 8 f16
        for (int i = tid; i < total; i += 256) dst[i] = src[i];
    }
#endif

    // ---- stage B operand (time-major -> fully contiguous 128B copies) ----
    {
        const int col  = tid >> 1;
        const int half = tid & 1;
        const int n    = n0 + col;
        const int t    = n & (TSEQ - 1);
        if (KTOT == 256) {
            uint4* dstv = (uint4*)&Xlds[(size_t)col * 256 + half * 128];
            if (half == 0) {                              // shifted rows K=0..127
                if (t >= d) {
                    const uint4* s = (const uint4*)(Xin + (size_t)(n - d) * 128);
                    #pragma unroll
                    for (int k = 0; k < 16; ++k) dstv[k] = s[k];
                } else {
                    uint4 z = {0u, 0u, 0u, 0u};
                    #pragma unroll
                    for (int k = 0; k < 16; ++k) dstv[k] = z;
                }
            } else {                                      // unshifted K=128..255
                const uint4* s = (const uint4*)(Xin + (size_t)n * 128);
                #pragma unroll
                for (int k = 0; k < 16; ++k) dstv[k] = s[k];
            }
        } else {                                          // KTOT==128 (1x1 conv)
            uint4* dstv = (uint4*)&Xlds[(size_t)col * 128 + half * 64];
            const uint4* s = (const uint4*)(Xin + (size_t)n * 128 + half * 64);
            #pragma unroll
            for (int k = 0; k < 8; ++k) dstv[k] = s[k];
        }
    }
    __syncthreads();

    const int lane = tid & 31;
    const int wave = tid >> 5;        // M block 0..7
    const int lhi  = lane >> 4;       // half-wave select
    const int llo  = lane & 15;

    v8f acc[8];
    {
        v8f z = {};
        #pragma unroll
        for (int nt = 0; nt < 8; ++nt) acc[nt] = z;
    }

    constexpr int KC = KTOT / 32;
    #pragma unroll
    for (int kc = 0; kc < KC; ++kc) {
        const v16h a = *(const v16h*)&Alds[(size_t)(wave * 16 + llo) * KTOT + kc * 32 + lhi * 16];
        #pragma unroll
        for (int nt = 0; nt < 8; ++nt) {
            const v16h bfrag =
                *(const v16h*)&Xlds[(size_t)(nt * 16 + llo) * KTOT + kc * 32 + lhi * 16];
            acc[nt] = __builtin_amdgcn_wmma_f32_16x16x32_f16(
                false, a, false, bfrag, (short)0, acc[nt], false, false);
        }
    }

    // ---- epilogue (8 consecutive channels per lane -> vector stores) ----
    float bv[8];
    #pragma unroll
    for (int r = 0; r < 8; ++r) bv[r] = bias[wave * 16 + lhi * 8 + r];

    #pragma unroll
    for (int nt = 0; nt < 8; ++nt) {
        const int    col  = n0 + nt * 16 + llo;
        const size_t base = (size_t)col * 128 + wave * 16 + lhi * 8;
        if (MODE == 0) {
            v8h y;
            #pragma unroll
            for (int r = 0; r < 8; ++r) y[r] = (_Float16)(acc[nt][r] + bv[r]);
            *(v8h*)(Yout16 + base) = y;
        } else {
            v8f hv = *(const v8f*)(Hres + base);
            v8h hx;
            #pragma unroll
            for (int r = 0; r < 8; ++r) {
                float hn = acc[nt][r] + bv[r] + hv[r];
                hv[r] = hn;
                hx[r] = (_Float16)hn;
            }
            *(v8f*)(Hres + base)   = hv;
            *(v8h*)(H16out + base) = hx;
            // total accumulates in the reference (B, C, T) layout
            const int bidx = col >> 13, tt = col & (TSEQ - 1);
            const size_t tb = ((size_t)bidx * 128 + wave * 16 + lhi * 8) * TSEQ + tt;
            #pragma unroll
            for (int r = 0; r < 8; ++r) Tot[tb + (size_t)r * TSEQ] += hv[r];
        }
    }
}

// ---------------------------------------------------------------------------
// Host side
// ---------------------------------------------------------------------------
extern "C" void kernel_launch(void* const* d_in, const int* in_sizes, int n_in,
                              void* d_out, int out_size, void* d_ws, size_t ws_size,
                              hipStream_t stream) {
    (void)in_sizes; (void)n_in; (void)ws_size;

    char* ws = (char*)d_ws;
    constexpr size_t SLOT    = 32768;                  // f16 elems per weight slot
    constexpr size_t OFF_H   = (size_t)8  << 20;       // f32 h          (32 MiB)
    constexpr size_t OFF_H16 = (size_t)40 << 20;       // f16 h          (16 MiB)
    constexpr size_t OFF_C1  = (size_t)56 << 20;       // f16 front act  ( 8 MiB)
    constexpr size_t OFF_F0  = (size_t)64 << 20;
    constexpr size_t OFF_F1  = (size_t)80 << 20;
    constexpr size_t OFF_G0  = (size_t)96 << 20;
    constexpr size_t OFF_G1  = (size_t)112 << 20;
    constexpr size_t OFF_Z   = (size_t)128 << 20;      // end ~144 MiB

    _Float16* wa    = (_Float16*)ws;
    float*    h     = (float*)   (ws + OFF_H);
    _Float16* h16   = (_Float16*)(ws + OFF_H16);
    _Float16* c1    = (_Float16*)(ws + OFF_C1);
    _Float16* fb[2] = {(_Float16*)(ws + OFF_F0), (_Float16*)(ws + OFF_F1)};
    _Float16* gb[2] = {(_Float16*)(ws + OFF_G0), (_Float16*)(ws + OFF_G1)};
    _Float16* z16   = (_Float16*)(ws + OFF_Z);
    float*    tot   = (float*)d_out;

    // d_out accumulates across blocks -> zero it every call (graph-safe)
    hipMemsetAsync(d_out, 0, (size_t)out_size * sizeof(float), stream);

    // input dict flatten: 0:x, 1-2:conv_ws, 3-4:conv_bs,
    // per block base=5+38b: f_ws[9], f_bs[9], g_ws[9], g_bs[9], scale_w, scale_b
    for (int b = 0; b < 4; ++b) {
        int base = 5 + b * 38;
        for (int i = 0; i < 9; ++i)
            prep_w2<<<64, 256, 0, stream>>>((const float*)d_in[base + i],
                                            wa + (size_t)(b * 19 + i) * SLOT);
        for (int i = 0; i < 9; ++i)
            prep_w2<<<64, 256, 0, stream>>>((const float*)d_in[base + 18 + i],
                                            wa + (size_t)(b * 19 + 9 + i) * SLOT);
        prep_w1<<<64, 256, 0, stream>>>((const float*)d_in[base + 36],
                                        wa + (size_t)(b * 19 + 18) * SLOT);
    }

    // ---- front convs ----
    front1_kernel<<<(64 * NTOT) / 256, 256, 0, stream>>>(
        (const float*)d_in[0], (const float*)d_in[1], (const float*)d_in[3], c1);
    front2_kernel<<<(128 * NTOT) / 256, 256, 0, stream>>>(
        (const float*)d_in[2], (const float*)d_in[4], c1, h, h16);

    constexpr int    GEMM_GRID = NTOT / 128;           // 512 workgroups
    constexpr size_t LDS256    = 2u * 128 * 256 * sizeof(_Float16); // 128 KiB
    constexpr size_t LDS128    = 2u * 128 * 128 * sizeof(_Float16); //  64 KiB

    // ---- gated residual blocks ----
    for (int b = 0; b < 4; ++b) {
        int base = 5 + b * 38;
        for (int i = 0; i < 9; ++i) {
            const _Float16* fin = (i == 0) ? h16 : fb[(i + 1) & 1];
            gemm_conv_wmma<256, 0><<<GEMM_GRID, 256, LDS256, stream>>>(
                wa + (size_t)(b * 19 + i) * SLOT, (const float*)d_in[base + 9 + i],
                fin, fb[i & 1], nullptr, nullptr, nullptr, 1 << i);
            const _Float16* gin = (i == 0) ? h16 : gb[(i + 1) & 1];
            gemm_conv_wmma<256, 0><<<GEMM_GRID, 256, LDS256, stream>>>(
                wa + (size_t)(b * 19 + 9 + i) * SLOT, (const float*)d_in[base + 27 + i],
                gin, gb[i & 1], nullptr, nullptr, nullptr, 1 << i);
        }
        gate_kernel<<<(128 * NTOT) / 256, 256, 0, stream>>>(fb[0], gb[0], z16);
        gemm_conv_wmma<128, 1><<<GEMM_GRID, 256, LDS128, stream>>>(
            wa + (size_t)(b * 19 + 18) * SLOT, (const float*)d_in[base + 37],
            z16, nullptr, h, tot, h16, 0);
    }
}
